// Attention_interaction_27376121544792
// MI455X (gfx1250) — matrix operand order repaired and checked by reading, not compiled, not run on hardware
//
#include <hip/hip_runtime.h>
#include <stdint.h>

// ---------------- problem constants ----------------
#define BHN   64                 // B*H
#define NSEQ  1024
#define DHEAD 64
#define WAVES 8
#define BRROW 128                // Q rows per block (8 waves * 16)
#define BCKEY 32                 // keys per KV iteration
#define NTILE (NSEQ / BCKEY)
#define KSTR  72                 // LDS stride (halves) for K tile (matches TDM padding)
#define VSTR  40                 // LDS stride (halves) for V^T tile
#define PSTR  72                 // LDS stride (halves) for per-wave P/O buffer

typedef _Float16 half16 __attribute__((ext_vector_type(16)));
typedef _Float16 half8  __attribute__((ext_vector_type(8)));
typedef _Float16 half4v __attribute__((ext_vector_type(4)));
typedef float    float8 __attribute__((ext_vector_type(8)));
typedef unsigned int u32x4 __attribute__((ext_vector_type(4)));
typedef int          i32x4 __attribute__((ext_vector_type(4)));
typedef int          i32x8 __attribute__((ext_vector_type(8)));

#if __has_include(<hip/amd_detail/amd_gfx1250_TDM.h>)
#define TDM_SIX_ARG 1
#endif

// ---- WMMA wrapper: D = A(16x32 f16) * B(32x16 f16) + C(16x16 f32) ----
__device__ __forceinline__ float8 wmma_f16(half16 a, half16 b, float8 c) {
  return __builtin_amdgcn_wmma_f32_16x16x32_f16(
      /*neg_a=*/false, a, /*neg_b=*/false, b,
      /*c_mod=*/(short)0, c, /*reuse_a=*/false, /*reuse_b=*/false);
}

// A-fragment: a[0..15] = p[0..15]   (p 16B aligned)
__device__ __forceinline__ half16 ld_a_frag(const _Float16* p) {
  half8 lo = *(const half8*)(p);
  half8 hi = *(const half8*)(p + 8);
  half16 r;
#pragma unroll
  for (int i = 0; i < 8; ++i) { r[i] = lo[i]; r[i + 8] = hi[i]; }
  return r;
}

// B-fragment: b[0..15] = p[0..15] (contiguous 32B, loaded as 2x16B)
__device__ __forceinline__ half16 ld_b_frag(const _Float16* p) {
  half8 lo = *(const half8*)(p);
  half8 hi = *(const half8*)(p + 8);
  half16 r;
#pragma unroll
  for (int i = 0; i < 8; ++i) { r[i] = lo[i]; r[i + 8] = hi[i]; }
  return r;
}

// ---- TDM: DMA one 32x64-half K tile (row stride 64 halves in memory) into LDS
// with a 72-half row stride, via pad_enable (pad 4 DWORDs every 32 DWORDs).
// D# bit packing per CDNA5 ISA 8.3/8.4. Issued by one wave; EXEC is ignored.
__device__ __forceinline__ void tdm_load_k_tile(const _Float16* gsrc, uint32_t lds_off) {
  const uint64_t ga = (uint64_t)(uintptr_t)gsrc;
  u32x4 g0;
  g0[0] = 1u;                                   // count=1 (valid), user mode
  g0[1] = lds_off;                              // lds_addr (bytes)
  g0[2] = (uint32_t)ga;                         // global_addr[31:0]
  g0[3] = (uint32_t)((ga >> 32) & 0x01FFFFFFu)  // global_addr[56:32]
        | (2u << 30);                           // type = 2 ("image")
  i32x8 g1;
  g1[0] = (1 << 16)                             // data_size = 1 -> 2 bytes
        | (1 << 20)                             // pad_enable
        | (4 << 22)                             // pad_interval: 32 DWORDs (one row)
        | (3 << 25);                            // pad_amount: 4 DWORDs (8 halves)
  g1[1] = (int)(64u << 16);                     // tensor_dim0 = 64 (bits 63:48)
  g1[2] = (int)((unsigned)BCKEY << 16);         // tensor_dim1 = 32 (bits 111:80 low)
  g1[3] = (int)(64u << 16);                     // tile_dim0 = 64  (bits 127:112)
  g1[4] = BCKEY;                                // tile_dim1 = 32  (bits 143:128)
  g1[5] = DHEAD;                                // tensor_dim0_stride = 64 (bits 207:160)
  g1[6] = 0;
  g1[7] = 0;
  i32x4 gz = {0, 0, 0, 0};
#if defined(TDM_SIX_ARG)
  i32x8 gz8 = {0, 0, 0, 0, 0, 0, 0, 0};
  __builtin_amdgcn_tensor_load_to_lds(g0, g1, gz, gz, gz8, 0);
#else
  __builtin_amdgcn_tensor_load_to_lds(g0, g1, gz, gz, 0);
#endif
}

// ---------------- prep 1: xn = l2norm(x) rows -> f16 ----------------
__global__ __launch_bounds__(256) void prep_norm_kernel(
    const float* __restrict__ x, _Float16* __restrict__ xn) {
  int t   = blockIdx.x * 256 + threadIdx.x;
  int rid = t >> 4;
  int c4  = (t & 15) * 4;
  const float4 v = *(const float4*)(x + (size_t)rid * DHEAD + c4);
  float ss = v.x * v.x + v.y * v.y + v.z * v.z + v.w * v.w;
  ss += __shfl_xor(ss, 1);
  ss += __shfl_xor(ss, 2);
  ss += __shfl_xor(ss, 4);
  ss += __shfl_xor(ss, 8);
  float rn = rsqrtf(ss + 1e-6f);
  half4v h;
  h[0] = (_Float16)(v.x * rn);
  h[1] = (_Float16)(v.y * rn);
  h[2] = (_Float16)(v.z * rn);
  h[3] = (_Float16)(v.w * rn);
  *(half4v*)(xn + (size_t)rid * DHEAD + c4) = h;
}

// ---------------- prep 2: y -> f16, W -> f16 ----------------
__global__ __launch_bounds__(256) void prep_cvt_kernel(
    const float* __restrict__ y, const float* __restrict__ w,
    _Float16* __restrict__ yh, _Float16* __restrict__ wh) {
  const size_t ny = (size_t)BHN * NSEQ * DHEAD;
  size_t i = ((size_t)blockIdx.x * 256 + threadIdx.x) * 4;
  if (i < ny) {
    float4 v = *(const float4*)(y + i);
    half4v h; h[0]=(_Float16)v.x; h[1]=(_Float16)v.y; h[2]=(_Float16)v.z; h[3]=(_Float16)v.w;
    *(half4v*)(yh + i) = h;
  } else {
    size_t j = i - ny;
    if (j < (size_t)DHEAD * DHEAD) {
      float4 v = *(const float4*)(w + j);
      half4v h; h[0]=(_Float16)v.x; h[1]=(_Float16)v.y; h[2]=(_Float16)v.z; h[3]=(_Float16)v.w;
      *(half4v*)(wh + j) = h;
    }
  }
}

// ---------------- main: flash attention + projection + blend ----------------
__global__ __launch_bounds__(256, 1) void attn_kernel(
    const _Float16* __restrict__ xn,   // normalized x, f16 [BH][N][64]
    const _Float16* __restrict__ yh,   // y f16
    const _Float16* __restrict__ wh,   // W f16 row-major [e][d]
    const float*    __restrict__ yf,   // y f32 (blend)
    const float*    __restrict__ pb,   // bias [64]
    const float*    __restrict__ gamma_p,
    const float*    __restrict__ sg0_p,
    const float*    __restrict__ sg1_p,
    float* __restrict__ out) {
  __shared__ _Float16 lK[2][BCKEY][KSTR];     // double-buffered key tiles (TDM target)
  __shared__ _Float16 lVT[2][DHEAD][VSTR];    // double-buffered transposed value tiles
  __shared__ _Float16 lP[WAVES][16][PSTR];    // per-wave P / O staging

  const int tid   = threadIdx.x;
  const int wave  = tid >> 5;
  const int lane  = tid & 31;
  const int lhalf = lane >> 4;     // 0: lanes 0-15, 1: lanes 16-31
  const int lm    = lane & 15;

  const int bh    = blockIdx.y;
  const int qrow0 = blockIdx.x * BRROW + wave * 16;

  const float qscale = 8.0f / gamma_p[0];    // 1/SCALE / gamma

  // Q A-fragments (K=0..31 and K=32..63), scale folded in
  const _Float16* qp = xn + ((size_t)(bh * NSEQ + qrow0 + lm)) * DHEAD;
  half16 aq0 = ld_a_frag(qp + 16 * lhalf);
  half16 aq1 = ld_a_frag(qp + 32 + 16 * lhalf);
  {
    _Float16 qs = (_Float16)qscale;
#pragma unroll
    for (int i = 0; i < 16; ++i) { aq0[i] *= qs; aq1[i] *= qs; }
  }

  float8 O0 = {}, O1 = {}, O2 = {}, O3 = {};
  float  lsum[8];
#pragma unroll
  for (int r = 0; r < 8; ++r) lsum[r] = 0.0f;

  // staging map: row = tid/8 (0..31), col block = (tid&7)*8
  const int sr = tid >> 3;
  const int sc = (tid & 7) * 8;
  const _Float16* xk_bh = xn + (size_t)bh * NSEQ * DHEAD;
  const _Float16* yv_bh = yh + (size_t)bh * NSEQ * DHEAD;

  const uint32_t lK_off[2] = { (uint32_t)(uintptr_t)&lK[0][0][0],
                               (uint32_t)(uintptr_t)&lK[1][0][0] };

  // ---- prologue: stage tile 0 (K via TDM from wave 0, V transposed by all) ----
  if (tid < 32) tdm_load_k_tile(xk_bh, lK_off[0]);
  {
    half8 vv = *(const half8*)(yv_bh + (size_t)sr * DHEAD + sc);
#pragma unroll
    for (int i = 0; i < 8; ++i) lVT[0][sc + i][sr] = vv[i];
  }
  __builtin_amdgcn_s_wait_tensorcnt(0);
  __syncthreads();

  for (int kt = 0; kt < NTILE; ++kt) {
    const int cur = kt & 1;
    const int nxt = cur ^ 1;
    const bool pf = (kt + 1) < NTILE;

    // ---- prefetch tile kt+1: TDM for K, global load of V rows (stores deferred) ----
    half8 vvn = {};
    if (pf) {
      if (tid < 32)
        tdm_load_k_tile(xk_bh + (size_t)(kt + 1) * BCKEY * DHEAD, lK_off[nxt]);
      vvn = *(const half8*)(yv_bh + (size_t)((kt + 1) * BCKEY + sr) * DHEAD + sc);
    }

    const _Float16 (*Kc)[KSTR] = lK[cur];
    const _Float16 (*Vc)[VSTR] = lVT[cur];

    // ---- S = Q * K^T (two 16x16 n-tiles), exp, stash P as f16 ----
#pragma unroll
    for (int j = 0; j < 2; ++j) {
      const _Float16* kp = &Kc[j * 16 + lm][16 * lhalf];
      half16 bk0 = ld_b_frag(kp);        // d = 0..31 chunk
      half16 bk1 = ld_b_frag(kp + 32);   // d = 32..63 chunk
      float8 S = {};
      S = wmma_f16(aq0, bk0, S);
      S = wmma_f16(aq1, bk1, S);
#pragma unroll
      for (int r = 0; r < 8; ++r) {
        // logits in [-0.8, 0.8] -> exp never overflows, no running max needed
        _Float16 ph = (_Float16)__expf(S[r]);
        lsum[r] += (float)ph;
        lP[wave][r + 8 * lhalf][j * 16 + lm] = ph;
      }
    }

    // ---- O += P * V ----
    half16 ap  = ld_a_frag(&lP[wave][lm][16 * lhalf]);          // P 16x32 A-frag
    half16 bv0 = ld_b_frag(&Vc[0 * 16 + lm][16 * lhalf]);
    half16 bv1 = ld_b_frag(&Vc[1 * 16 + lm][16 * lhalf]);
    half16 bv2 = ld_b_frag(&Vc[2 * 16 + lm][16 * lhalf]);
    half16 bv3 = ld_b_frag(&Vc[3 * 16 + lm][16 * lhalf]);
    O0 = wmma_f16(ap, bv0, O0);
    O1 = wmma_f16(ap, bv1, O1);
    O2 = wmma_f16(ap, bv2, O2);
    O3 = wmma_f16(ap, bv3, O3);

    // ---- land prefetched V into the other buffer ----
    if (pf) {
#pragma unroll
      for (int i = 0; i < 8; ++i) lVT[nxt][sc + i][sr] = vvn[i];
    }

    __builtin_amdgcn_s_wait_tensorcnt(0);
    __syncthreads();
  }

  // ---- row sums: reduce across the 16 lanes of each half-wave ----
#pragma unroll
  for (int r = 0; r < 8; ++r) {
    float v = lsum[r];
    v += __shfl_xor(v, 1);
    v += __shfl_xor(v, 2);
    v += __shfl_xor(v, 4);
    v += __shfl_xor(v, 8);
    lsum[r] = 1.0f / v;
  }

  // ---- normalized O -> LDS (f16), reload in A-layout ----
#pragma unroll
  for (int r = 0; r < 8; ++r) {
    int row = r + 8 * lhalf;
    lP[wave][row][ 0 + lm] = (_Float16)(O0[r] * lsum[r]);
    lP[wave][row][16 + lm] = (_Float16)(O1[r] * lsum[r]);
    lP[wave][row][32 + lm] = (_Float16)(O2[r] * lsum[r]);
    lP[wave][row][48 + lm] = (_Float16)(O3[r] * lsum[r]);
  }
  half16 ao0 = ld_a_frag(&lP[wave][lm][16 * lhalf]);
  half16 ao1 = ld_a_frag(&lP[wave][lm][32 + 16 * lhalf]);

  const float g0 = __expf(sg0_p[0]);
  const float g1 = __expf(sg1_p[0]);
  const float c0 = g0 / (g0 + g1);
  const float c1 = g1 / (g0 + g1);

  // ---- projection (O @ W^T + b) and blend with y ----
#pragma unroll
  for (int eb = 0; eb < 4; ++eb) {
    const _Float16* wr = wh + (size_t)(eb * 16 + lm) * DHEAD + 16 * lhalf;
    half16 bw0 = ld_b_frag(wr);        // d = 0..31
    half16 bw1 = ld_b_frag(wr + 32);   // d = 32..63
    float8 R = {};
    R = wmma_f16(ao0, bw0, R);
    R = wmma_f16(ao1, bw1, R);
    float bcol = pb[eb * 16 + lm];
#pragma unroll
    for (int r = 0; r < 8; ++r) {
      int row = qrow0 + r + 8 * lhalf;
      size_t idx = ((size_t)(bh * NSEQ + row)) * DHEAD + eb * 16 + lm;
      out[idx] = c0 * yf[idx] + c1 * (R[r] + bcol);
    }
  }
}

// ---------------- host launcher ----------------
extern "C" void kernel_launch(void* const* d_in, const int* in_sizes, int n_in,
                              void* d_out, int out_size, void* d_ws, size_t ws_size,
                              hipStream_t stream) {
  const float* x    = (const float*)d_in[0];
  const float* y    = (const float*)d_in[1];
  const float* pw   = (const float*)d_in[2];
  const float* pbv  = (const float*)d_in[3];
  const float* gam  = (const float*)d_in[4];
  const float* sg0  = (const float*)d_in[5];
  const float* sg1  = (const float*)d_in[6];
  float* out = (float*)d_out;

  const size_t nel = (size_t)BHN * NSEQ * DHEAD;   // 4,194,304
  _Float16* xn = (_Float16*)d_ws;
  _Float16* yh = xn + nel;
  _Float16* wh = yh + nel;

  prep_norm_kernel<<<(nel / DHEAD) * 16 / 256, 256, 0, stream>>>(x, xn);
  prep_cvt_kernel<<<(unsigned)((nel + (size_t)DHEAD * DHEAD) / 1024), 256, 0, stream>>>(y, pw, yh, wh);
  attn_kernel<<<dim3(NSEQ / BRROW, BHN), 256, 0, stream>>>(
      xn, yh, wh, y, pbv, gam, sg0, sg1, out);
}
